// QACILSTM_55147380081296
// MI455X (gfx1250) — matrix operand order repaired and checked by reading, not compiled, not run on hardware
//
#include <hip/hip_runtime.h>
#include <hip/hip_bf16.h>
#include <math.h>

// ---------------------------------------------------------------------------
// Problem constants (from reference): T=128, B=64, H=768, NH=12, dh=64
// ---------------------------------------------------------------------------
#define TT 128
#define BB 64
#define HH 768
#define NHEAD 12
#define DH 64
#define H4 3072   // 4*H
#define H2 1536   // 2*H
#define QKVN 2304 // 3*H

typedef __bf16 bf16_t;
typedef __attribute__((ext_vector_type(16))) __bf16 v16bf;
typedef __attribute__((ext_vector_type(8)))  float  v8f;

union FragBF { v16bf v; uint4 q[2]; };

static __device__ __forceinline__ float sigmoidf_(float x) {
    return 1.0f / (1.0f + __expf(-x));
}

// ---------------------------------------------------------------------------
// gfx1250 data movers (guarded; degrade gracefully if a builtin is missing)
// ---------------------------------------------------------------------------
#define ENABLE_TDM   1
#define ENABLE_ASYNC 1

#if ENABLE_TDM && __has_builtin(__builtin_amdgcn_tensor_load_to_lds)
#define HAVE_TDM 1
typedef unsigned int u32x4 __attribute__((ext_vector_type(4)));
typedef int          i32x8 __attribute__((ext_vector_type(8)));
typedef int          i32x4 __attribute__((ext_vector_type(4)));

// TDM: load a 64-row x 32-col bf16 tile (row stride = K elements) from global
// into LDS at lds_off, padding each 16-DWORD row with 4 DWORDs (-> stride 40
// bf16 elements, matching the padded LDS panel layout).  ISA 08 / D# groups.
static __device__ __forceinline__ void tdm_load_A(const bf16_t* gsrc,
                                                  unsigned lds_off, int K) {
    unsigned long long ga = (unsigned long long)(size_t)gsrc;
    u32x4 g0;
    g0[0] = 1u;                                   // count=1, user descriptor
    g0[1] = lds_off;                              // lds_addr
    g0[2] = (unsigned)(ga & 0xFFFFFFFFu);         // global_addr[31:0]
    g0[3] = (unsigned)((ga >> 32) & 0x01FFFFFFu)  // global_addr[56:32]
          | (2u << 30);                           // type=2 ("image")
    unsigned td0 = 0x40000000u, td1 = 0x40000000u; // huge tensor dims (no clip)
    i32x8 g1;
    g1[0] = (1 << 16)        // data_size = 2 bytes (bf16)
          | (1 << 20)        // pad_enable
          | (3 << 22)        // pad_interval code 3 -> every 16 DWORDs
          | (3 << 25);       // pad_amount  code 3 -> 4 DWORDs
    g1[1] = (int)((td0 & 0xFFFFu) << 16);                    // tensor_dim0 lo16
    g1[2] = (int)((td0 >> 16) | ((td1 & 0xFFFFu) << 16));    // dim0 hi / dim1 lo
    g1[3] = (int)((td1 >> 16) | (32u << 16));                // dim1 hi, tile_dim0=32
    g1[4] = 64;                                              // tile_dim1=64, tile_dim2=0
    g1[5] = (int)(unsigned)K;                                // tensor_dim0_stride lo32
    g1[6] = 0;                                               // stride hi, dim1_stride lo
    g1[7] = 0;
    i32x4 z4 = {0, 0, 0, 0};
#if __clang_major__ >= 23
    i32x8 z8 = {0, 0, 0, 0, 0, 0, 0, 0};
    __builtin_amdgcn_tensor_load_to_lds(g0, g1, z4, z4, z8, 0);
#else
    __builtin_amdgcn_tensor_load_to_lds(g0, g1, z4, z4, 0);
#endif
}
static __device__ __forceinline__ void wait_tensor0() {
#if __has_builtin(__builtin_amdgcn_s_wait_tensorcnt)
    __builtin_amdgcn_s_wait_tensorcnt(0);
#else
    asm volatile("s_wait_tensorcnt 0x0" ::: "memory");
#endif
    asm volatile("" ::: "memory");
}
#elif ENABLE_ASYNC && __has_builtin(__builtin_amdgcn_global_load_async_to_lds_b128)
#define HAVE_ASYNC 1
typedef __attribute__((address_space(1))) char glb_char;
typedef __attribute__((address_space(3))) char lds_char;
static __device__ __forceinline__ void async_copy16(void* lds, const void* glb) {
    __builtin_amdgcn_global_load_async_to_lds_b128(
        (glb_char*)(unsigned long long)(size_t)glb,
        (lds_char*)(unsigned)(size_t)lds, 0, 0);
}
static __device__ __forceinline__ void wait_async0() {
#if __has_builtin(__builtin_amdgcn_s_wait_asynccnt)
    __builtin_amdgcn_s_wait_asynccnt(0);
#else
    asm volatile("s_wait_asynccnt 0x0" ::: "memory");
#endif
    asm volatile("" ::: "memory");
}
#endif

// ---------------------------------------------------------------------------
// Generic bf16 WMMA GEMM:  C[M,N] = A[M,K] * B[K,N]  (row-major bf16)
// Block = 256 thr = 8 waves; block tile 64 x TNB (TNB = 64 or 128).
// A panel staged by the TDM engine (async, s_wait_tensorcnt); B panel is
// software-pipelined through registers so global latency overlaps compute.
// All NT B fragments are loaded before the NT back-to-back WMMAs.
// ---------------------------------------------------------------------------
#define LDSTR 40  // padded LDS stride (elements) for the k=32 panels

template <int TNB>
__global__ __launch_bounds__(256)
void gemm_bf16_wmma(const bf16_t* __restrict__ A, const bf16_t* __restrict__ Bm,
                    int M, int N, int K,
                    const float* __restrict__ bias,
                    float* __restrict__ Cf, int ldcf,
                    bf16_t* __restrict__ Cb0, int ldc0,
                    bf16_t* __restrict__ Cb1, int ldc1)
{
    constexpr int NT  = TNB / 32;   // C tiles per wave (2 or 4)
    constexpr int NCH = TNB / 64;   // B register chunks per thread (1 or 2)

    __shared__ __align__(16) bf16_t sA [64  * LDSTR];  // A panel (m, k)
    __shared__ __align__(16) bf16_t sBt[TNB * LDSTR];  // B panel transposed (n, k)

    const int tid  = threadIdx.x;
    const int lane = tid & 31;
    const int wave = tid >> 5;          // 0..7
    const int bm   = blockIdx.y * 64;
    const int bn   = blockIdx.x * TNB;

    const int wm   = wave & 3;          // wave's 16-row tile
    const int wcp  = wave >> 2;         // 0/1 -> first/second half of tile cols

    const int mloc = lane & 15;
    const int hi   = lane >> 4;         // lane half (ISA bf16 fragment layout)

    v8f acc[NT];
#pragma unroll
    for (int tc = 0; tc < NT; ++tc) acc[tc] = v8f{};

    // staging coordinates (constant over the K loop)
    const int arow = tid >> 2;          // 0..63
    const int acol = (tid & 3) * 8;     // 0,8,16,24
    int bkk[NCH], bnn[NCH];
#pragma unroll
    for (int it = 0; it < NCH; ++it) {
        const int idx = tid + it * 256;          // 0 .. 32*TNB/8 - 1
        bkk[it] = idx / (TNB / 8);               // 0..31
        bnn[it] = (idx % (TNB / 8)) * 8;         // col within block tile
    }

    // ---- pipeline prologue: B panel k0=0 into registers ----
    uint4 breg[NCH];
#pragma unroll
    for (int it = 0; it < NCH; ++it)
        breg[it] = *(const uint4*)(Bm + (size_t)bkk[it] * N + bn + bnn[it]);

    for (int k0 = 0; k0 < K; k0 += 32) {
        // ---- stage A panel 64x32 (async TDM / async-LDS / sync) ----
#if defined(HAVE_TDM)
        if (wave == 0)
            tdm_load_A(A + (size_t)bm * K + k0, (unsigned)(size_t)&sA[0], K);
#elif defined(HAVE_ASYNC)
        async_copy16(&sA[arow * LDSTR + acol],
                     A + (size_t)(bm + arow) * K + k0 + acol);
#else
        *(uint4*)(&sA[arow * LDSTR + acol]) =
            *(const uint4*)(A + (size_t)(bm + arow) * K + k0 + acol);
#endif
        // ---- scatter the register-staged B panel into LDS (transposed) ----
#pragma unroll
        for (int it = 0; it < NCH; ++it) {
            const bf16_t* tv = (const bf16_t*)&breg[it];
#pragma unroll
            for (int e = 0; e < 8; ++e)
                sBt[(bnn[it] + e) * LDSTR + bkk[it]] = tv[e];
        }
        // ---- pipeline: issue global loads for the NEXT B panel now, so the
        //      memory latency overlaps this iteration's WMMA work ----
        if (k0 + 32 < K) {
#pragma unroll
            for (int it = 0; it < NCH; ++it)
                breg[it] = *(const uint4*)(Bm + (size_t)(k0 + 32 + bkk[it]) * N
                                              + bn + bnn[it]);
#if !defined(HAVE_TDM)
            __builtin_prefetch(A + (size_t)(bm + arow) * K + k0 + 32 + acol, 0, 3);
#endif
        }
#if defined(HAVE_TDM)
        wait_tensor0();
#elif defined(HAVE_ASYNC)
        wait_async0();
#endif
        __syncthreads();

        // ---- A fragment (16x32 bf16, ISA 7.12.2 layout) ----
        FragBF af;
        {
            const bf16_t* ap = &sA[(wm * 16 + mloc) * LDSTR];
            af.q[0] = *(const uint4*)(ap + hi * 8);        // k = hi*8 .. +7
            af.q[1] = *(const uint4*)(ap + 16 + hi * 8);   // k = 16+hi*8 .. +7
        }
        // ---- load ALL B fragments first, then back-to-back WMMAs ----
        FragBF bfv[NT];
#pragma unroll
        for (int tc = 0; tc < NT; ++tc) {
            const int tn = wcp * NT + tc;
            const bf16_t* bp = &sBt[(tn * 16 + mloc) * LDSTR + hi * 8];
            bfv[tc].q[0] = *(const uint4*)(bp);
            bfv[tc].q[1] = *(const uint4*)(bp + 16);
        }
#pragma unroll
        for (int tc = 0; tc < NT; ++tc)
            acc[tc] = __builtin_amdgcn_wmma_f32_16x16x32_bf16(
                          false, af.v, false, bfv[tc].v, (short)0, acc[tc],
                          false, false);
        __syncthreads();
    }

    // ---- epilogue: VGPR j -> M = hi*8 + j, N = lane%16 ----
#pragma unroll
    for (int tc = 0; tc < NT; ++tc) {
        const int n = bn + (wcp * NT + tc) * 16 + mloc;
        if (n >= N) continue;
        const float bv = bias ? bias[n] : 0.0f;
#pragma unroll
        for (int j = 0; j < 8; ++j) {
            const int m = bm + wm * 16 + hi * 8 + j;
            if (m >= M) continue;
            const float v = acc[tc][j] + bv;
            if (Cf)  Cf [(size_t)m * ldcf + n] = v;
            if (Cb0) Cb0[(size_t)m * ldc0 + n] = (bf16_t)v;
            if (Cb1) Cb1[(size_t)m * ldc1 + n] = (bf16_t)v;
        }
    }
}

// ---------------------------------------------------------------------------
// Per-step gather: lr[b*2+0] = left, lr[b*2+1] = right (bf16 rows of buf)
// ---------------------------------------------------------------------------
__global__ __launch_bounds__(256)
void gather_kernel(const bf16_t* __restrict__ buf, const int* __restrict__ ping,
                   int t, bf16_t* __restrict__ lr)
{
    int idx = blockIdx.x * 256 + threadIdx.x;           // B*H threads
    if (idx >= BB * HH) return;
    int b = idx / HH, h = idx % HH;
    int pt = ping[b * TT + t];
    bf16_t right = buf[((size_t)t * BB + b) * HH + h];
    bf16_t left = right;
    if (pt > 0) {
        int row = pt - 1;
        if (row > TT - 1) row = TT - 1;
        left = buf[((size_t)row * BB + b) * HH + h];
    }
    lr[(size_t)(b * 2 + 0) * HH + h] = left;
    lr[(size_t)(b * 2 + 1) * HH + h] = right;
}

// ---------------------------------------------------------------------------
// Attention: one wave per (b, head); 4x4 scores via wave32 shfl_xor reduce.
// ---------------------------------------------------------------------------
__global__ __launch_bounds__(256)
void attn_kernel(const float* __restrict__ qkv_qa, const float* __restrict__ qkv_lr,
                 const float* __restrict__ bq, const float* __restrict__ bk,
                 const float* __restrict__ bv, bf16_t* __restrict__ ctx)
{
    const int wv   = blockIdx.x * 8 + (threadIdx.x >> 5);
    const int lane = threadIdx.x & 31;
    const int b    = wv / NHEAD;
    const int head = wv % NHEAD;
    if (b >= BB) return;

    const float scale = 0.125f;  // 1/sqrt(64)
    const int hbase = head * DH;
    const int d0 = lane, d1 = lane + 32;

    float q[4][2], k[4][2], v[4][2];
#pragma unroll
    for (int tok = 0; tok < 4; ++tok) {
        const float* src = (tok < 2)
            ? (qkv_qa + (size_t)(b * 2 + tok)     * QKVN)
            : (qkv_lr + (size_t)(b * 2 + tok - 2) * QKVN);
        q[tok][0] = src[hbase + d0]        + bq[hbase + d0];
        q[tok][1] = src[hbase + d1]        + bq[hbase + d1];
        k[tok][0] = src[HH + hbase + d0]   + bk[hbase + d0];
        k[tok][1] = src[HH + hbase + d1]   + bk[hbase + d1];
        v[tok][0] = src[2*HH + hbase + d0] + bv[hbase + d0];
        v[tok][1] = src[2*HH + hbase + d1] + bv[hbase + d1];
    }

    float sc[4][4];
#pragma unroll
    for (int i = 0; i < 4; ++i)
#pragma unroll
        for (int j = 0; j < 4; ++j)
            sc[i][j] = q[i][0] * k[j][0] + q[i][1] * k[j][1];

#pragma unroll
    for (int off = 16; off > 0; off >>= 1)
#pragma unroll
        for (int i = 0; i < 4; ++i)
#pragma unroll
            for (int j = 0; j < 4; ++j)
                sc[i][j] += __shfl_xor(sc[i][j], off, 32);

    float p[4][4];
#pragma unroll
    for (int i = 0; i < 4; ++i) {
        float mx = -1e30f;
#pragma unroll
        for (int j = 0; j < 4; ++j) { sc[i][j] *= scale; mx = fmaxf(mx, sc[i][j]); }
        float sum = 0.0f;
#pragma unroll
        for (int j = 0; j < 4; ++j) { p[i][j] = __expf(sc[i][j] - mx); sum += p[i][j]; }
        float inv = 1.0f / sum;
#pragma unroll
        for (int j = 0; j < 4; ++j) p[i][j] *= inv;
    }

#pragma unroll
    for (int tok = 0; tok < 4; ++tok) {
        float c0 = 0.0f, c1 = 0.0f;
#pragma unroll
        for (int j = 0; j < 4; ++j) { c0 += p[tok][j] * v[j][0]; c1 += p[tok][j] * v[j][1]; }
        ctx[(size_t)b * H4 + tok * HH + hbase + d0] = (bf16_t)c0;
        ctx[(size_t)b * H4 + tok * HH + hbase + d1] = (bf16_t)c1;
    }
}

// ---------------------------------------------------------------------------
// LSTM elementwise
// ---------------------------------------------------------------------------
__global__ __launch_bounds__(256)
void lstm_kernel(const float* __restrict__ gates,
                 const float* __restrict__ b_ih, const float* __restrict__ b_hh,
                 float* __restrict__ hst, float* __restrict__ cst,
                 bf16_t* __restrict__ cat_h, float* __restrict__ hs_out)
{
    int idx = blockIdx.x * 256 + threadIdx.x;  // B*H
    if (idx >= BB * HH) return;
    int b = idx / HH, h = idx % HH;
    const float* g = gates + (size_t)b * H4;
    float ig = g[h]          + b_ih[h]          + b_hh[h];
    float fg = g[HH + h]     + b_ih[HH + h]     + b_hh[HH + h];
    float gg = g[2 * HH + h] + b_ih[2 * HH + h] + b_hh[2 * HH + h];
    float og = g[3 * HH + h] + b_ih[3 * HH + h] + b_hh[3 * HH + h];
    float c  = sigmoidf_(fg) * cst[idx] + sigmoidf_(ig) * tanhf(gg);
    float hh2 = sigmoidf_(og) * tanhf(c);
    cst[idx] = c;
    hst[idx] = hh2;
    cat_h[(size_t)b * H2 + HH + h] = (bf16_t)hh2;
    hs_out[idx] = hh2;
}

// ---------------------------------------------------------------------------
// One-time prep kernels (fp32 -> bf16 packing)
// ---------------------------------------------------------------------------
__global__ void cvt_f32_bf16(const float* __restrict__ s, bf16_t* __restrict__ d, int n) {
    int i = blockIdx.x * 256 + threadIdx.x;
    if (i < n) d[i] = (bf16_t)s[i];
}
__global__ void pack_qkv(const float* __restrict__ Wq, const float* __restrict__ Wk,
                         const float* __restrict__ Wv, bf16_t* __restrict__ Wqkv) {
    int i = blockIdx.x * 256 + threadIdx.x;           // HH*QKVN
    if (i >= HH * QKVN) return;
    int k = i / QKVN, n = i % QKVN;
    float val = (n < HH)     ? Wq[(size_t)k * HH + n]
              : (n < 2 * HH) ? Wk[(size_t)k * HH + (n - HH)]
                             : Wv[(size_t)k * HH + (n - 2 * HH)];
    Wqkv[i] = (bf16_t)val;
}
__global__ void pack_catw(const float* __restrict__ Wih, const float* __restrict__ Whh,
                          bf16_t* __restrict__ Wcat) {
    int i = blockIdx.x * 256 + threadIdx.x;           // H2*H4
    if (i >= H2 * H4) return;
    int k = i / H4, n = i % H4;
    float val = (k < HH) ? Wih[(size_t)k * H4 + n] : Whh[(size_t)(k - HH) * H4 + n];
    Wcat[i] = (bf16_t)val;
}
__global__ void pack_qa(const float* __restrict__ q0, const float* __restrict__ a0,
                        bf16_t* __restrict__ qa) {
    int i = blockIdx.x * 256 + threadIdx.x;           // B*H
    if (i >= BB * HH) return;
    int b = i / HH, h = i % HH;
    qa[(size_t)(b * 2 + 0) * HH + h] = (bf16_t)q0[i];
    qa[(size_t)(b * 2 + 1) * HH + h] = (bf16_t)a0[i];
}
__global__ void init_hcat(const float* __restrict__ h0, bf16_t* __restrict__ cat_h) {
    int i = blockIdx.x * 256 + threadIdx.x;           // B*H
    if (i >= BB * HH) return;
    int b = i / HH, h = i % HH;
    cat_h[(size_t)b * H2 + HH + h] = (bf16_t)h0[i];
}

// ---------------------------------------------------------------------------
// Host driver
// ---------------------------------------------------------------------------
extern "C" void kernel_launch(void* const* d_in, const int* in_sizes, int n_in,
                              void* d_out, int out_size, void* d_ws, size_t ws_size,
                              hipStream_t stream) {
    const float* question = (const float*)d_in[0];
    const float* answer   = (const float*)d_in[1];
    const float* x        = (const float*)d_in[2];
    const int*   ping     = (const int*)  d_in[3];
    const float* h0       = (const float*)d_in[4];
    const float* c0       = (const float*)d_in[5];
    const float* Wq       = (const float*)d_in[6];
    const float* bq       = (const float*)d_in[7];
    const float* Wk       = (const float*)d_in[8];
    const float* bk       = (const float*)d_in[9];
    const float* Wv       = (const float*)d_in[10];
    const float* bv       = (const float*)d_in[11];
    const float* W_attn   = (const float*)d_in[12];
    const float* b_attn   = (const float*)d_in[13];
    const float* W_ih     = (const float*)d_in[14];
    const float* b_ih     = (const float*)d_in[15];
    const float* W_hh     = (const float*)d_in[16];
    const float* b_hh     = (const float*)d_in[17];
    float* out = (float*)d_out;

    // workspace carve-up (all sizes multiples of 256B)
    char* ws = (char*)d_ws;
    size_t off = 0;
    auto carve = [&](size_t bytes) { void* p = ws + off; off += (bytes + 255) & ~(size_t)255; return p; };

    bf16_t* buf_bf  = (bf16_t*)carve((size_t)TT * BB * HH * 2);  // mutable token buffer
    bf16_t* Wqkv    = (bf16_t*)carve((size_t)HH * QKVN * 2);
    bf16_t* Wattn_b = (bf16_t*)carve((size_t)H4 * HH * 2);
    bf16_t* Wcat    = (bf16_t*)carve((size_t)H2 * H4 * 2);
    bf16_t* qa_bf   = (bf16_t*)carve((size_t)2 * BB * HH * 2);
    float*  qkv_qa  = (float*) carve((size_t)2 * BB * QKVN * 4);
    bf16_t* lr_bf   = (bf16_t*)carve((size_t)2 * BB * HH * 2);
    float*  qkv_lr  = (float*) carve((size_t)2 * BB * QKVN * 4);
    bf16_t* ctx_bf  = (bf16_t*)carve((size_t)BB * H4 * 2);
    bf16_t* cat_bf  = (bf16_t*)carve((size_t)BB * H2 * 2);
    float*  gates   = (float*) carve((size_t)BB * H4 * 4);
    float*  h_st    = (float*) carve((size_t)BB * HH * 4);
    float*  c_st    = (float*) carve((size_t)BB * HH * 4);
    (void)ws_size; (void)n_in; (void)in_sizes; (void)out_size;

    const int BH_BLK = (BB * HH + 255) / 256;  // 192

    // ---- one-time prep ----
    pack_qkv <<<(HH * QKVN + 255) / 256, 256, 0, stream>>>(Wq, Wk, Wv, Wqkv);
    cvt_f32_bf16<<<(H4 * HH + 255) / 256, 256, 0, stream>>>(W_attn, Wattn_b, H4 * HH);
    pack_catw<<<(H2 * H4 + 255) / 256, 256, 0, stream>>>(W_ih, W_hh, Wcat);
    cvt_f32_bf16<<<(TT * BB * HH + 255) / 256, 256, 0, stream>>>(x, buf_bf, TT * BB * HH);
    pack_qa  <<<BH_BLK, 256, 0, stream>>>(question, answer, qa_bf);
    init_hcat<<<BH_BLK, 256, 0, stream>>>(h0, cat_bf);
    hipMemcpyAsync(h_st, h0, (size_t)BB * HH * 4, hipMemcpyDeviceToDevice, stream);
    hipMemcpyAsync(c_st, c0, (size_t)BB * HH * 4, hipMemcpyDeviceToDevice, stream);

    // hoisted: QKV projection of the constant q0/a0 tokens (M=128, N=2304, K=768)
    gemm_bf16_wmma<128><<<dim3(QKVN / 128, 2), 256, 0, stream>>>(
        qa_bf, Wqkv, 2 * BB, QKVN, HH,
        nullptr, qkv_qa, QKVN, nullptr, 0, nullptr, 0);

    // ---- the sequential scan ----
    for (int t = 0; t < TT; ++t) {
        gather_kernel<<<BH_BLK, 256, 0, stream>>>(buf_bf, ping, t, lr_bf);

        // left/right QKV projection: (128 x 768) * (768 x 2304)
        gemm_bf16_wmma<128><<<dim3(QKVN / 128, 2), 256, 0, stream>>>(
            lr_bf, Wqkv, 2 * BB, QKVN, HH,
            nullptr, qkv_lr, QKVN, nullptr, 0, nullptr, 0);

        // 4-token attention per (b, head)
        attn_kernel<<<(BB * NHEAD) / 8, 256, 0, stream>>>(
            qkv_qa, qkv_lr, bq, bk, bv, ctx_bf);

        // attention output proj: (64 x 3072) * (3072 x 768) + b_attn
        //   -> bf16 into buf[t] AND into cat[:, :768]
        gemm_bf16_wmma<64><<<dim3(HH / 64, 1), 256, 0, stream>>>(
            ctx_bf, Wattn_b, BB, HH, H4,
            b_attn, nullptr, 0,
            buf_bf + (size_t)t * BB * HH, HH,
            cat_bf, H2);

        // fused LSTM gates: (64 x 1536) * (1536 x 3072)
        gemm_bf16_wmma<128><<<dim3(H4 / 128, 1), 256, 0, stream>>>(
            cat_bf, Wcat, BB, H4, H2,
            nullptr, gates, H4, nullptr, 0, nullptr, 0);

        // elementwise LSTM + write hs[t]
        lstm_kernel<<<BH_BLK, 256, 0, stream>>>(
            gates, b_ih, b_hh, h_st, c_st, cat_bf,
            out + (size_t)t * BB * HH);
    }

    // final (h, c) outputs
    hipMemcpyAsync(out + (size_t)TT * BB * HH,           h_st, (size_t)BB * HH * 4,
                   hipMemcpyDeviceToDevice, stream);
    hipMemcpyAsync(out + (size_t)TT * BB * HH + BB * HH, c_st, (size_t)BB * HH * 4,
                   hipMemcpyDeviceToDevice, stream);
}
